// SINGLE_Encoder_14637248544931
// MI455X (gfx1250) — matrix operand order repaired and checked by reading, not compiled, not run on hardware
//
#include <hip/hip_runtime.h>

// ---------------- types ----------------
typedef __attribute__((ext_vector_type(16))) _Float16 v16h;
typedef __attribute__((ext_vector_type(8)))  _Float16 v8h;
typedef __attribute__((ext_vector_type(8)))  float    v8f;

#define Bc   64
#define Sv   201
#define Sp   208      // S padded to 16
#define Ec   128
#define Hc   8
#define Dc   16
#define FFc  512
#define Lc   6
#define Kv   224      // S padded to multiple of 32 (attention K-dim)
#define MT   13       // ceil(201/16)

#define GF_BIAS 1
#define GF_RELU 2
#define GF_F32  4
#define GF_F16  8

union FragU { v16h v; v8h h[2]; unsigned int u[8]; };

// A fragment (16-bit 16x32): lane -> row M=lane&15; elements 0..15 = K koff..koff+15,
// koff = (lane<16 ? 0 : 16). Two 16B loads.
__device__ __forceinline__ v16h load_a32(const _Float16* p, int lane) {
  int koff = (lane & 16) ? 16 : 0;
  FragU u;
  u.h[0] = *(const v8h*)(p + koff);
  u.h[1] = *(const v8h*)(p + koff + 8);
  return u.v;
}
// B fragment (16-bit 32x16): lane -> col N=lane&15; elements 0..15 = contiguous K run
// (lanes 0-15: K 0..15, lanes 16-31: K 16..31).  p must point at that 16-half run.
__device__ __forceinline__ v16h load_b32(const _Float16* p) {
  FragU u;
  u.h[0] = *(const v8h*)(p);
  u.h[1] = *(const v8h*)(p + 8);
  return u.v;
}

// ------------- weight repack: row-major f32 [K,N] -> WMMA-B tile-lane-major f16 ------
// block (kt,nt) = 512 halves: dst[li*16 + e] = W[kt*32 + (li<16?0:16) + e][nt*16 + (li&15)]
__global__ __launch_bounds__(256) void repack_w(const float* __restrict__ src,
                                                _Float16* __restrict__ dst, int K, int N) {
  int t = blockIdx.x * 256 + threadIdx.x;
  if (t >= K * N) return;
  int blk = t >> 9, rem = t & 511;
  int li = rem >> 4, e = rem & 15;
  int ntiles = N >> 4;
  int kt = blk / ntiles, nt = blk - kt * ntiles;
  int n = (nt << 4) + (li & 15);
  int k = (kt << 5) + ((li & 16) ? 16 : 0) + e;
  dst[t] = (_Float16)src[(size_t)k * N + n];
}

// ------------- initial embeddings ------------------------------------------------
__global__ __launch_bounds__(256) void embed_kernel(
    const float* __restrict__ depot, const float* __restrict__ nodes,
    const float* __restrict__ Wd, const float* __restrict__ bd,
    const float* __restrict__ Wn, const float* __restrict__ bn,
    float* __restrict__ X, _Float16* __restrict__ Xh) {
  int idx = blockIdx.x * 256 + threadIdx.x;
  if (idx >= Bc * Sv * Ec) return;
  int e = idx % Ec; int s = (idx / Ec) % Sv; int b = idx / (Ec * Sv);
  float v;
  if (s == 0) {
    const float* d = depot + b * 3;
    v = bd[e] + d[0] * Wd[e] + d[1] * Wd[Ec + e] + d[2] * Wd[2 * Ec + e];
  } else {
    const float* nd = nodes + ((size_t)b * 200 + (s - 1)) * 4;
    v = bn[e] + nd[0] * Wn[e] + nd[1] * Wn[Ec + e] + nd[2] * Wn[2 * Ec + e] + nd[3] * Wn[3 * Ec + e];
  }
  size_t o = ((size_t)b * Sp + s) * Ec + e;
  X[o] = v; Xh[o] = (_Float16)v;
}

// ------------- generic batched GEMM: C[b] = A[b] (201xK f16) @ Wrep (KxN) ---------
// one wave computes a 16x64 C slab (4 WMMAs per K-step), K multiple of 32
__global__ __launch_bounds__(128) void gemm_f16(
    const _Float16* __restrict__ A, const _Float16* __restrict__ W,
    const float* __restrict__ bias, float* __restrict__ Cf,
    _Float16* __restrict__ Ch, int K, int N, int flags) {
  int wid = (blockIdx.x * 128 + threadIdx.x) >> 5;
  int lane = threadIdx.x & 31;
  int n64 = N >> 6;
  int per_b = MT * n64;
  if (wid >= Bc * per_b) return;
  int b = wid / per_b;
  int r = wid - b * per_b;
  int mt = r / n64;
  int nb = r - mt * n64;

  v8f acc[4];
  const v8f z = {0.f, 0.f, 0.f, 0.f, 0.f, 0.f, 0.f, 0.f};
  acc[0] = z; acc[1] = z; acc[2] = z; acc[3] = z;

  const _Float16* Arow = A + ((size_t)b * Sp + mt * 16 + (lane & 15)) * K;
  int kt32 = K >> 5, ntiles = N >> 4;
  for (int kt = 0; kt < kt32; ++kt) {
    v16h a = load_a32(Arow + kt * 32, lane);
    const _Float16* wb = W + ((size_t)kt * ntiles + (nb << 2)) * 512 + lane * 16;
#pragma unroll
    for (int t = 0; t < 4; ++t) {
      v16h bf = load_b32(wb + t * 512);
      acc[t] = __builtin_amdgcn_wmma_f32_16x16x32_f16(false, a, false, bf,
                                                      (short)0, acc[t], false, false);
    }
  }
  int rbase = mt * 16 + ((lane & 16) ? 8 : 0);
#pragma unroll
  for (int t = 0; t < 4; ++t) {
    int n = (nb << 6) + t * 16 + (lane & 15);
    float bv = (flags & GF_BIAS) ? bias[n] : 0.0f;
#pragma unroll
    for (int g = 0; g < 8; ++g) {
      int row = rbase + g;
      if (row < Sv) {
        float v = acc[t][g] + bv;
        if (flags & GF_RELU) v = fmaxf(v, 0.0f);
        size_t idx = ((size_t)b * Sp + row) * N + n;
        if (flags & GF_F32) Cf[idx] = v;
        if (flags & GF_F16) Ch[idx] = (_Float16)v;
      }
    }
  }
}

// ------------- V GEMM with per-head transposed epilogue: vhT[b,h,d,s] -------------
__global__ __launch_bounds__(128) void gemm_v_f16(
    const _Float16* __restrict__ A, const _Float16* __restrict__ W,
    _Float16* __restrict__ vhT) {
  int wid = (blockIdx.x * 128 + threadIdx.x) >> 5;
  int lane = threadIdx.x & 31;
  const int n64 = 2, per_b = MT * n64;
  if (wid >= Bc * per_b) return;
  int b = wid / per_b;
  int r = wid - b * per_b;
  int mt = r / n64, nb = r - mt * n64;

  v8f acc[4];
  const v8f z = {0.f, 0.f, 0.f, 0.f, 0.f, 0.f, 0.f, 0.f};
  acc[0] = z; acc[1] = z; acc[2] = z; acc[3] = z;
  const _Float16* Arow = A + ((size_t)b * Sp + mt * 16 + (lane & 15)) * Ec;
  for (int kt = 0; kt < 4; ++kt) {
    v16h a = load_a32(Arow + kt * 32, lane);
    const _Float16* wb = W + ((size_t)kt * 8 + (nb << 2)) * 512 + lane * 16;
#pragma unroll
    for (int t = 0; t < 4; ++t) {
      v16h bf = load_b32(wb + t * 512);
      acc[t] = __builtin_amdgcn_wmma_f32_16x16x32_f16(false, a, false, bf,
                                                      (short)0, acc[t], false, false);
    }
  }
  int rbase = mt * 16 + ((lane & 16) ? 8 : 0);
#pragma unroll
  for (int t = 0; t < 4; ++t) {
    int n = (nb << 6) + t * 16 + (lane & 15);
    int h = n >> 4, d = n & 15;
    _Float16* vp = vhT + ((size_t)(b * Hc + h) * Dc + d) * Kv;
#pragma unroll
    for (int g = 0; g < 8; ++g) {
      int row = rbase + g;
      if (row < Sv) vp[row] = (_Float16)acc[t][g];
    }
  }
}

// ------------- score = q . k^T per head (K=16 zero-padded to 32) ------------------
__global__ __launch_bounds__(128) void score_f16(
    const _Float16* __restrict__ qh, const _Float16* __restrict__ kh,
    float* __restrict__ sc) {
  int wid = (blockIdx.x * 128 + threadIdx.x) >> 5;
  int lane = threadIdx.x & 31;
  const int per_h = MT * MT;
  if (wid >= Bc * Hc * per_h) return;
  int b = wid / (per_h * Hc);
  int h = (wid / per_h) % Hc;
  int r = wid % per_h;
  int mt = r / MT, nt = r - mt * MT;

  int rq = mt * 16 + (lane & 15);
  FragU ua;
  if (lane < 16) {
    const _Float16* qp = qh + ((size_t)b * Sp + rq) * Ec + h * Dc;
    ua.h[0] = *(const v8h*)(qp);
    ua.h[1] = *(const v8h*)(qp + 8);
  } else {
#pragma unroll
    for (int i = 0; i < 8; ++i) ua.u[i] = 0;   // K=16..31 pad
  }

  FragU ub;
  if (lane < 16) {
    int rk = nt * 16 + lane;
    const _Float16* kp = kh + ((size_t)b * Sp + rk) * Ec + h * Dc;
    ub.h[0] = *(const v8h*)(kp);
    ub.h[1] = *(const v8h*)(kp + 8);
  } else {
#pragma unroll
    for (int i = 0; i < 8; ++i) ub.u[i] = 0;   // K=16..31 pad
  }

  v8f c = {0.f, 0.f, 0.f, 0.f, 0.f, 0.f, 0.f, 0.f};
  c = __builtin_amdgcn_wmma_f32_16x16x32_f16(false, ua.v, false, ub.v, (short)0, c, false, false);

  int m = nt * 16 + (lane & 15);
  if (m < Sv) {
    float* sp = sc + ((size_t)(b * Hc + h) * Sv) * Sv + m;
    int rbase = mt * 16 + ((lane & 16) ? 8 : 0);
#pragma unroll
    for (int g = 0; g < 8; ++g) {
      int n = rbase + g;
      if (n < Sv) sp[(size_t)n * Sv] = c[g];
    }
  }
}

// ------------- score-aggregation MLP (per (b,n,m), in-place on score) -------------
__global__ __launch_bounds__(256) void mlp_kernel(
    float* __restrict__ sc, const float* __restrict__ route,
    const float* __restrict__ w1, const float* __restrict__ b1,
    const float* __restrict__ w2, const float* __restrict__ b2) {
  __shared__ float sW1[256], sW2[128], sB1[16], sB2[8];
  int tid = threadIdx.x;
  sW1[tid] = w1[tid];
  if (tid < 128) sW2[tid] = w2[tid];
  if (tid < 16) sB1[tid] = b1[tid];
  if (tid < 8) sB2[tid] = b2[tid];
  __syncthreads();
  int idx = blockIdx.x * 256 + tid;
  if (idx >= Bc * Sv * Sv) return;
  int b = idx / (Sv * Sv);
  int r = idx - b * (Sv * Sv);
  int n = r / Sv, m = r - n * Sv;
  size_t base = ((size_t)(b * Hc) * Sv + n) * Sv + m;
  const size_t hs = (size_t)Sv * Sv;
  float cin[16];
#pragma unroll
  for (int h = 0; h < 8; ++h) {
    cin[h]     = sc[base + h * hs];
    cin[8 + h] = route[base + h * hs];
  }
  float h1[16];
#pragma unroll
  for (int j = 0; j < 16; ++j) {
    float s = sB1[j];
#pragma unroll
    for (int i = 0; i < 16; ++i) s += cin[i] * sW1[i * 16 + j];
    h1[j] = fmaxf(s, 0.0f);
  }
#pragma unroll
  for (int h = 0; h < 8; ++h) {
    float s = sB2[h];
#pragma unroll
    for (int j = 0; j < 16; ++j) s += h1[j] * sW2[j * 8 + h];
    sc[base + h * hs] = s;
  }
}

// ------------- wave-per-row softmax -> f16 probs, zero pad columns to Kv ----------
__global__ __launch_bounds__(128) void softmax_f16(
    const float* __restrict__ sc, _Float16* __restrict__ wh) {
  int wid = (blockIdx.x * 128 + threadIdx.x) >> 5;
  int lane = threadIdx.x & 31;
  if (wid >= Bc * Hc * Sv) return;
  int n = wid % Sv;
  int h = (wid / Sv) % Hc;
  int b = wid / (Sv * Hc);
  const float* rowp = sc + ((size_t)(b * Hc + h) * Sv + n) * Sv;
  float vals[7];
  float mx = -3.4e38f;
#pragma unroll
  for (int i = 0; i < 7; ++i) {
    int m = lane + i * 32;
    vals[i] = (m < Sv) ? rowp[m] : -3.4e38f;
    mx = fmaxf(mx, vals[i]);
  }
  for (int off = 16; off > 0; off >>= 1) mx = fmaxf(mx, __shfl_xor(mx, off, 32));
  float sum = 0.f;
#pragma unroll
  for (int i = 0; i < 7; ++i) {
    int m = lane + i * 32;
    float e = (m < Sv) ? __expf(vals[i] - mx) : 0.f;
    vals[i] = e; sum += e;
  }
  for (int off = 16; off > 0; off >>= 1) sum += __shfl_xor(sum, off, 32);
  float inv = 1.0f / sum;
  _Float16* wrow = wh + ((size_t)(b * Hc + h) * Sp + n) * Kv;
#pragma unroll
  for (int i = 0; i < 7; ++i) {
    int m = lane + i * 32;
    if (m < Sv) wrow[m] = (_Float16)(vals[i] * inv);
  }
  if (lane < (Kv - Sv)) wrow[Sv + lane] = (_Float16)0.0f;
}

// ------------- attention apply: o = probs @ v  (M=S, K=224, N=16 per head) --------
__global__ __launch_bounds__(128) void attnout_f16(
    const _Float16* __restrict__ wh, const _Float16* __restrict__ vhT,
    _Float16* __restrict__ oh) {
  int wid = (blockIdx.x * 128 + threadIdx.x) >> 5;
  int lane = threadIdx.x & 31;
  if (wid >= Bc * Hc * MT) return;
  int b = wid / (Hc * MT);
  int r = wid - b * (Hc * MT);
  int h = r / MT, mt = r - h * MT;
  v8f acc = {0.f, 0.f, 0.f, 0.f, 0.f, 0.f, 0.f, 0.f};
  const _Float16* Arow = wh + ((size_t)(b * Hc + h) * Sp + mt * 16 + (lane & 15)) * Kv;
  const _Float16* Brow = vhT + ((size_t)(b * Hc + h) * Dc + (lane & 15)) * Kv + ((lane & 16) ? 16 : 0);
#pragma unroll
  for (int kt = 0; kt < 7; ++kt) {
    v16h a = load_a32(Arow + kt * 32, lane);
    v16h bf = load_b32(Brow + kt * 32);
    acc = __builtin_amdgcn_wmma_f32_16x16x32_f16(false, a, false, bf, (short)0, acc, false, false);
  }
  int d = lane & 15;
  int rbase = mt * 16 + ((lane & 16) ? 8 : 0);
#pragma unroll
  for (int g = 0; g < 8; ++g) {
    int s = rbase + g;
    if (s < Sv) oh[((size_t)b * Sp + s) * Ec + h * Dc + d] = (_Float16)acc[g];
  }
}

// ------------- InstanceNorm1d over sequence axis: out = IN(X + R)*w + b ----------
__global__ __launch_bounds__(128) void inorm_kernel(
    const float* __restrict__ X, const float* __restrict__ R,
    const float* __restrict__ w, const float* __restrict__ bb,
    float* __restrict__ Of, _Float16* __restrict__ Oh) {
  int b = blockIdx.x, e = threadIdx.x;
  size_t base = (size_t)b * Sp * Ec + e;
  float s = 0.f, s2 = 0.f;
  for (int i = 0; i < Sv; ++i) {
    float t = X[base + (size_t)i * Ec] + R[base + (size_t)i * Ec];
    s += t; s2 += t * t;
  }
  float mean = s * (1.0f / Sv);
  float var = s2 * (1.0f / Sv) - mean * mean;
  float inv = rsqrtf(var + 1e-5f);
  float ww = w[e], bv = bb[e];
  for (int i = 0; i < Sv; ++i) {
    size_t o = base + (size_t)i * Ec;
    float t = X[o] + R[o];
    float y = (t - mean) * inv * ww + bv;
    Of[o] = y; Oh[o] = (_Float16)y;
  }
}

// ------------- pad-row zeroing for vhT (s = 201..223) ----------------------------
__global__ __launch_bounds__(256) void vpad_zero(_Float16* __restrict__ vhT) {
  const int PAD = Kv - Sv;
  int idx = blockIdx.x * 256 + threadIdx.x;
  if (idx >= Bc * Hc * Dc * PAD) return;
  int j = idx % PAD, rd = idx / PAD;
  vhT[(size_t)rd * Kv + Sv + j] = (_Float16)0.0f;
}

__global__ __launch_bounds__(256) void copy_out_kernel(const float* __restrict__ X,
                                                       float* __restrict__ out) {
  int idx = blockIdx.x * 256 + threadIdx.x;
  if (idx >= Bc * Sv * Ec) return;
  int e = idx % Ec; int s = (idx / Ec) % Sv; int b = idx / (Ec * Sv);
  out[idx] = X[((size_t)b * Sp + s) * Ec + e];
}

// =================================================================================
extern "C" void kernel_launch(void* const* d_in, const int* in_sizes, int n_in,
                              void* d_out, int out_size, void* d_ws, size_t ws_size,
                              hipStream_t stream) {
  (void)in_sizes; (void)n_in; (void)out_size; (void)ws_size;
  const float* depot = (const float*)d_in[0];
  const float* nodes = (const float*)d_in[1];
  const float* route = (const float*)d_in[2];
  const float* Wd  = (const float*)d_in[3];
  const float* bd  = (const float*)d_in[4];
  const float* Wn  = (const float*)d_in[5];
  const float* bn  = (const float*)d_in[6];
  const float* Wq  = (const float*)d_in[7];
  const float* Wk  = (const float*)d_in[8];
  const float* Wv  = (const float*)d_in[9];
  const float* Wo  = (const float*)d_in[10];
  const float* bo  = (const float*)d_in[11];
  const float* A1w = (const float*)d_in[12];
  const float* A1b = (const float*)d_in[13];
  const float* A2w = (const float*)d_in[14];
  const float* A2b = (const float*)d_in[15];
  const float* n1w = (const float*)d_in[16];
  const float* n1b = (const float*)d_in[17];
  const float* n2w = (const float*)d_in[18];
  const float* n2b = (const float*)d_in[19];
  const float* F1w = (const float*)d_in[20];
  const float* F1b = (const float*)d_in[21];
  const float* F2w = (const float*)d_in[22];
  const float* F2b = (const float*)d_in[23];

  char* ws = (char*)d_ws;
  size_t off = 0;
  auto take = [&](size_t bytes) -> void* {
    void* p = (void*)(ws + off);
    off += (bytes + 255) & ~(size_t)255;
    return p;
  };
  float*    x    = (float*)take((size_t)Bc * Sp * Ec * 4);
  _Float16* xh   = (_Float16*)take((size_t)Bc * Sp * Ec * 2);
  _Float16* qh   = (_Float16*)take((size_t)Bc * Sp * Ec * 2);
  _Float16* kh   = (_Float16*)take((size_t)Bc * Sp * Ec * 2);
  _Float16* vhT  = (_Float16*)take((size_t)Bc * Hc * Dc * Kv * 2);
  _Float16* oh   = (_Float16*)take((size_t)Bc * Sp * Ec * 2);
  float*    o1   = (float*)take((size_t)Bc * Sp * Ec * 4);
  _Float16* o1h  = (_Float16*)take((size_t)Bc * Sp * Ec * 2);
  float*    mh   = (float*)take((size_t)Bc * Sp * Ec * 4);   // shared mh / ff2 buffer
  _Float16* ffh  = (_Float16*)take((size_t)Bc * Sp * FFc * 2);
  float*    sc   = (float*)take((size_t)Bc * Hc * Sv * Sv * 4);
  _Float16* whb  = (_Float16*)take((size_t)Bc * Hc * Sp * Kv * 2);
  _Float16* WqR  = (_Float16*)take((size_t)Lc * Ec * Ec * 2);
  _Float16* WkR  = (_Float16*)take((size_t)Lc * Ec * Ec * 2);
  _Float16* WvR  = (_Float16*)take((size_t)Lc * Ec * Ec * 2);
  _Float16* WoR  = (_Float16*)take((size_t)Lc * Ec * Ec * 2);
  _Float16* F1R  = (_Float16*)take((size_t)Lc * Ec * FFc * 2);
  _Float16* F2R  = (_Float16*)take((size_t)Lc * FFc * Ec * 2);

  for (int l = 0; l < Lc; ++l) {
    repack_w<<<64, 256, 0, stream>>>(Wq + (size_t)l * Ec * Ec, WqR + (size_t)l * Ec * Ec, Ec, Ec);
    repack_w<<<64, 256, 0, stream>>>(Wk + (size_t)l * Ec * Ec, WkR + (size_t)l * Ec * Ec, Ec, Ec);
    repack_w<<<64, 256, 0, stream>>>(Wv + (size_t)l * Ec * Ec, WvR + (size_t)l * Ec * Ec, Ec, Ec);
    repack_w<<<64, 256, 0, stream>>>(Wo + (size_t)l * Ec * Ec, WoR + (size_t)l * Ec * Ec, Ec, Ec);
    repack_w<<<256, 256, 0, stream>>>(F1w + (size_t)l * Ec * FFc, F1R + (size_t)l * Ec * FFc, Ec, FFc);
    repack_w<<<256, 256, 0, stream>>>(F2w + (size_t)l * FFc * Ec, F2R + (size_t)l * FFc * Ec, FFc, Ec);
  }
  vpad_zero<<<(Bc * Hc * Dc * (Kv - Sv) + 255) / 256, 256, 0, stream>>>(vhT);
  embed_kernel<<<(Bc * Sv * Ec + 255) / 256, 256, 0, stream>>>(depot, nodes, Wd, bd, Wn, bn, x, xh);

  const int gN128 = (Bc * MT * 2 + 3) / 4;   // waves/4, N=128
  const int gN512 = (Bc * MT * 8 + 3) / 4;   // N=512
  const int gScore = (Bc * Hc * MT * MT + 3) / 4;
  const int gSoft  = (Bc * Hc * Sv + 3) / 4;
  const int gAttn  = (Bc * Hc * MT + 3) / 4;
  const int gMlp   = (Bc * Sv * Sv + 255) / 256;

  for (int l = 0; l < Lc; ++l) {
    gemm_f16<<<gN128, 128, 0, stream>>>(xh, WqR + (size_t)l * Ec * Ec, nullptr, nullptr, qh, Ec, Ec, GF_F16);
    gemm_f16<<<gN128, 128, 0, stream>>>(xh, WkR + (size_t)l * Ec * Ec, nullptr, nullptr, kh, Ec, Ec, GF_F16);
    gemm_v_f16<<<gN128, 128, 0, stream>>>(xh, WvR + (size_t)l * Ec * Ec, vhT);
    score_f16<<<gScore, 128, 0, stream>>>(qh, kh, sc);
    mlp_kernel<<<gMlp, 256, 0, stream>>>(sc, route, A1w + (size_t)l * 256, A1b + (size_t)l * 16,
                                         A2w + (size_t)l * 128, A2b + (size_t)l * 8);
    softmax_f16<<<gSoft, 128, 0, stream>>>(sc, whb);
    attnout_f16<<<gAttn, 128, 0, stream>>>(whb, vhT, oh);
    gemm_f16<<<gN128, 128, 0, stream>>>(oh, WoR + (size_t)l * Ec * Ec, bo + (size_t)l * Ec,
                                        mh, nullptr, Ec, Ec, GF_BIAS | GF_F32);
    inorm_kernel<<<Bc, 128, 0, stream>>>(x, mh, n1w + (size_t)l * Ec, n1b + (size_t)l * Ec, o1, o1h);
    gemm_f16<<<gN512, 128, 0, stream>>>(o1h, F1R + (size_t)l * Ec * FFc, F1b + (size_t)l * FFc,
                                        nullptr, ffh, Ec, FFc, GF_BIAS | GF_RELU | GF_F16);
    gemm_f16<<<gN128, 128, 0, stream>>>(ffh, F2R + (size_t)l * FFc * Ec, F2b + (size_t)l * Ec,
                                        mh, nullptr, FFc, Ec, GF_BIAS | GF_F32);
    inorm_kernel<<<Bc, 128, 0, stream>>>(o1, mh, n2w + (size_t)l * Ec, n2b + (size_t)l * Ec, x, xh);
  }
  copy_out_kernel<<<(Bc * Sv * Ec + 255) / 256, 256, 0, stream>>>(x, (float*)d_out);
}
